// HumorModel_60971355734382
// MI455X (gfx1250) — matrix operand order repaired and checked by reading, not compiled, not run on hardware
//
#include <hip/hip_runtime.h>

// ---------------------------------------------------------------------------
// CDNA5 WMMA (wave32): D(16x16 f32) = A(16x32 bf16) x B(32x16 bf16) + C
// ---------------------------------------------------------------------------
typedef __bf16 v16bf __attribute__((ext_vector_type(16)));
typedef __bf16 bf8   __attribute__((ext_vector_type(8)));
typedef float  v8f   __attribute__((ext_vector_type(8)));

union Frag16 { v16bf v; bf8 h[2]; };

#define T_DIM   256
#define B_DIM   256
#define M_DIM   512
#define NROWS   65280        // (T-1)*B
#define XLD     544          // x_in feature stride, 518 padded to mult of 32
#define LDSW    520          // LDS hidden-act row stride (260 dw % 64 banks = 4 -> conflict-free)
#define LDS_BYTES (64 * LDSW * 2)

__device__ __forceinline__ v8f v8zero() {
  v8f z;
#pragma unroll
  for (int i = 0; i < 8; ++i) z[i] = 0.0f;
  return z;
}

__device__ __forceinline__ v8f wmma_bf16(v16bf a, v16bf b, v8f c) {
  // (neg_a, A, neg_b, B, c_mod, C, reuse_a, reuse_b)
  return __builtin_amdgcn_wmma_f32_16x16x32_bf16(false, a, false, b, (short)0, c, false, false);
}

// A fragment (16x32): lanes 0-15 hold K {0..7,16..23} of row M=lane,
// lanes 16-31 hold K {8..15,24..31} of row M=lane-16. `p` is per-lane base.
__device__ __forceinline__ void load_frag_a(Frag16& a, const __bf16* p, int koff) {
  a.h[0] = *(const bf8*)(p + koff);
  a.h[1] = *(const bf8*)(p + koff + 16);
}
// B fragment (32x16) from Wt[out, Kpad]: lane holds column N=lane&15,
// K {0..7,16..23} (lanes 0-15) or {8..15,24..31} (lanes 16-31). `p` is per-lane base.
__device__ __forceinline__ void load_frag_b(Frag16& b, const __bf16* p, int koff) {
  b.h[0] = *(const bf8*)(p + koff);
  b.h[1] = *(const bf8*)(p + koff + 16);
}

// Explicit two-phase software-pipelined GEMM segment.
// Wave computes 32 rows x 64 cols: A blocks pa0/pa1 (16 rows each, shared B),
// 4 B N-tiles at stride LDW. KSTEPS K-tiles of 32. kofsB = K offset in weights.
// The kt+1 load in the last iteration fetches one dead (never consumed) K-tile;
// all such addresses stay inside the workspace (global) or LDS range-check.
template <int LDW, int KSTEPS>
__device__ __forceinline__ void gemm_pipe(const __bf16* pa0, const __bf16* pa1,
                                          const __bf16* pb, int kofsB,
                                          v8f* acc0, v8f* acc1) {
  Frag16 a0[2], a1[2], bb[2][4];
  auto loadk = [&](int kt, int ph) {
    const int ka = kt * 32;
    load_frag_a(a0[ph], pa0, ka);
    load_frag_a(a1[ph], pa1, ka);
#pragma unroll
    for (int nt = 0; nt < 4; ++nt)
      load_frag_b(bb[ph][nt], pb, nt * 16 * LDW + kofsB + ka);
  };
  loadk(0, 0);
#pragma unroll 2
  for (int kt = 0; kt < KSTEPS; ++kt) {
    const int ph = kt & 1;
    loadk(kt + 1, ph ^ 1);          // prefetch next K-tile into the other phase
#pragma unroll
    for (int nt = 0; nt < 4; ++nt) {
      acc0[nt] = wmma_bf16(a0[ph].v, bb[ph][nt].v, acc0[nt]);
      acc1[nt] = wmma_bf16(a1[ph].v, bb[ph][nt].v, acc1[nt]);
    }
  }
}

// bias + relu + bf16 -> LDS (row-major, stride LDSW); two 16-row blocks,
// wave covers 32 rows x 64 cols (4 N-tiles).
__device__ __forceinline__ void write_act2(const v8f* a0, const v8f* a1,
                                           const float* __restrict__ bias,
                                           __bf16* __restrict__ H,
                                           int rb, int ch, int lane) {
  const int nl = lane & 15;
  const int mh = (lane < 16) ? 0 : 8;
  const int r0 = rb * 32 + mh;
#pragma unroll
  for (int nt = 0; nt < 4; ++nt) {
    const int col = ch * 64 + nt * 16 + nl;
    const float bc = bias[col];
#pragma unroll
    for (int v = 0; v < 8; ++v) {
      float y0 = a0[nt][v] + bc; y0 = y0 > 0.0f ? y0 : 0.0f;
      float y1 = a1[nt][v] + bc; y1 = y1 > 0.0f ? y1 : 0.0f;
      H[(r0 + v) * LDSW + col]      = (__bf16)y0;
      H[(r0 + 16 + v) * LDSW + col] = (__bf16)y1;
    }
  }
}

// ---------------------------------------------------------------------------
// Fused 5-layer MLP. 64-row tile per block, 16 waves (512 threads);
// wave = 32 rows x 64 cols (two 16-row A blocks share every B fragment;
// acc = 2x4x8 = 64 VGPRs). Layer0 = two K-segment passes (primary X, K=17*32;
// secondary X2, K=K2*32 at weight K-offset 544).
// ---------------------------------------------------------------------------
template <int LDX2, int K2, int LDW0>
__global__ __launch_bounds__(512)
void fused_mlp_kernel(const __bf16* __restrict__ X,  int xRowOff,
                      const __bf16* __restrict__ X2, int x2RowOff,
                      const __bf16* __restrict__ W0,
                      const __bf16* __restrict__ W1,
                      const __bf16* __restrict__ W2,
                      const __bf16* __restrict__ W3,
                      const __bf16* __restrict__ W4,
                      const float* __restrict__ B0, const float* __restrict__ B1,
                      const float* __restrict__ B2, const float* __restrict__ B3,
                      const float* __restrict__ B4,
                      float* __restrict__ Y, int ldy, int headTiles, int headCols) {
  extern __shared__ __bf16 hbuf[];   // 64 x LDSW bf16 (dynamic: 66,560 B)

  const int lane = threadIdx.x & 31;
  const int w    = threadIdx.x >> 5;
  const int rb   = w >> 3;           // 0..1  : 32-row block
  const int ch   = w & 7;            // 0..7  : 64-col eighth
  const int l    = lane & 15;
  const int hi   = lane >> 4;
  const int rowTile = blockIdx.x * 64;

  v8f acc0[4], acc1[4];
#pragma unroll
  for (int nt = 0; nt < 4; ++nt) { acc0[nt] = v8zero(); acc1[nt] = v8zero(); }

  // ---- layer 0 : A from global x_in (bf16), two K segments
  {
    const __bf16* pb  = W0 + (ch * 64 + l) * LDW0 + hi * 8;  // B base (N=l, Khalf=hi)
    const __bf16* pa0 = X  + (rowTile + xRowOff  + rb * 32 + l) * XLD  + hi * 8;
    const __bf16* pa1 = pa0 + 16 * XLD;
    const __bf16* qa0 = X2 + (rowTile + x2RowOff + rb * 32 + l) * LDX2 + hi * 8;
    const __bf16* qa1 = qa0 + 16 * LDX2;
    __builtin_prefetch((const void*)pa0, 0, 1);   // global_prefetch_b8 of streamed rows
    __builtin_prefetch((const void*)qa0, 0, 1);
    gemm_pipe<LDW0, 17>(pa0, pa1, pb, 0,   acc0, acc1);
    gemm_pipe<LDW0, K2>(qa0, qa1, pb, 544, acc0, acc1);
  }
  write_act2(acc0, acc1, B0, hbuf, rb, ch, lane);
  __syncthreads();

  // ---- layers 1..3 : 512x512, A from LDS, in-place with barriers
  const __bf16* Ws[3] = { W1, W2, W3 };
  const float*  Bs[3] = { B1, B2, B3 };
  const __bf16* pl0 = hbuf + (rb * 32 + l) * LDSW + hi * 8;
  const __bf16* pl1 = pl0 + 16 * LDSW;
  for (int ly = 0; ly < 3; ++ly) {
#pragma unroll
    for (int nt = 0; nt < 4; ++nt) { acc0[nt] = v8zero(); acc1[nt] = v8zero(); }
    const __bf16* pb = Ws[ly] + (ch * 64 + l) * 512 + hi * 8;
    gemm_pipe<512, 16>(pl0, pl1, pb, 0, acc0, acc1);
    __syncthreads();                 // everyone done reading hbuf
    write_act2(acc0, acc1, Bs[ly], hbuf, rb, ch, lane);
    __syncthreads();                 // writes visible
  }

  // ---- head : 512 -> headTiles*16 cols (no relu), f32 to global
  for (int nt = ch; nt < headTiles; nt += 8) {   // wave-uniform bounds
    v8f c0 = v8zero(), c1 = v8zero();
    const __bf16* pb = W4 + (nt * 16 + l) * 512 + hi * 8;
    for (int kt = 0; kt < 16; ++kt) {
      Frag16 a0, a1, b;
      load_frag_a(a0, pl0, kt * 32);
      load_frag_a(a1, pl1, kt * 32);
      load_frag_b(b, pb, kt * 32);
      c0 = wmma_bf16(a0.v, b.v, c0);
      c1 = wmma_bf16(a1.v, b.v, c1);
    }
    const int col = nt * 16 + (lane & 15);
    if (col < headCols) {
      const float bc = B4[col];
      const int mh = (lane < 16) ? 0 : 8;
      const int r0 = rowTile + rb * 32 + mh;
#pragma unroll
      for (int v = 0; v < 8; ++v) {
        Y[(r0 + v) * ldy + col]      = c0[v] + bc;
        Y[(r0 + 16 + v) * ldy + col] = c1[v] + bc;
      }
    }
  }
}

// ---------------------------------------------------------------------------
// Weight prep: f32 W[K, cols] -> bf16 Wt[dstRows, ldDst] transposed + K-padded
// ---------------------------------------------------------------------------
__global__ void prep_weight_kernel(const float* __restrict__ src, __bf16* __restrict__ dst,
                                   int validK, int srcCols, int dstRows,
                                   int ldDst, int kseg, int koffDst, int srcRowOff) {
  long long i = (long long)blockIdx.x * blockDim.x + threadIdx.x;
  long long total = (long long)dstRows * kseg;
  if (i >= total) return;
  int k = (int)(i % kseg);
  int o = (int)(i / kseg);
  float v = 0.0f;
  if (k < validK && o < srcCols) v = src[(long long)(srcRowOff + k) * srcCols + o];
  dst[(long long)o * ldDst + koffDst + k] = (__bf16)v;
}

// ---------------------------------------------------------------------------
// Preprocess: heading-frame transform -> x_in cols 0..5 (bf16) + zero pad cols
// ---------------------------------------------------------------------------
__global__ void preprocess_kernel(const float* __restrict__ orient_q,
                                  const float* __restrict__ trans,
                                  __bf16* __restrict__ xin, int n) {
  int r = blockIdx.x * blockDim.x + threadIdx.x;
  if (r >= n) return;
  const float* q  = orient_q + (long long)r * 4;
  const float* tr = trans    + (long long)r * 3;
  float nrm = sqrtf(q[0]*q[0] + q[1]*q[1] + q[2]*q[2] + q[3]*q[3]);
  float inv = 1.0f / fmaxf(nrm, 1e-12f);
  float qw = q[0]*inv, qx = q[1]*inv, qy = q[2]*inv, qz = q[3]*inv;
  float nh = sqrtf(qw*qw + qz*qz + 1e-12f);
  float hw = qw / nh, hz = -qz / nh;            // inv_heading = (hw,0,0,hz)
  float hqw = hw*qw - hz*qz;
  float hqx = hw*qx - hz*qy;
  float hqy = hw*qy + hz*qx;
  float hqz = hw*qz + hz*qw;
  float t0 = -2.0f * hz * tr[1];
  float t1 =  2.0f * hz * tr[0];
  float ht0 = tr[0] + hw*t0 - hz*t1;
  float ht1 = tr[1] + hw*t1 + hz*t0;
  float ht2 = tr[2];
  float nn = sqrtf(hqx*hqx + hqy*hqy + hqz*hqz);
  float angle = 2.0f * atan2f(nn, hqw);
  float scale = (nn < 1e-6f) ? 2.0f : angle / fmaxf(nn, 1e-12f);
  __bf16* row = xin + (long long)r * XLD;
  row[0] = (__bf16)ht0; row[1] = (__bf16)ht1; row[2] = (__bf16)ht2;
  row[3] = (__bf16)(scale * hqx);
  row[4] = (__bf16)(scale * hqy);
  row[5] = (__bf16)(scale * hqz);
  for (int c = 518; c < XLD; ++c) row[c] = (__bf16)0.0f;   // K padding
}

__global__ void cast_motion_kernel(const float* __restrict__ m, __bf16* __restrict__ xin,
                                   long long total) {
  long long i = (long long)blockIdx.x * blockDim.x + threadIdx.x;
  if (i >= total) return;
  long long r = i >> 9;              // / 512
  int c = (int)(i & 511);
  xin[r * XLD + 6 + c] = (__bf16)m[i];
}

// z = qm + exp(0.5*qv)*eps  -> bf16, stride 64 (cols 48..63 zero)
__global__ void zsamp_kernel(const float* __restrict__ qout, const float* __restrict__ eps,
                             __bf16* __restrict__ zbuf, long long total) {
  long long i = (long long)blockIdx.x * blockDim.x + threadIdx.x;
  if (i >= total) return;
  long long r = i >> 6;
  int c = (int)(i & 63);
  float v = 0.0f;
  if (c < 48) {
    float qm = qout[r * 96 + c];
    float qv = qout[r * 96 + 48 + c];
    v = qm + expf(0.5f * qv) * eps[r * 48 + c];
  }
  zbuf[i] = (__bf16)v;
}

// ---------------------------------------------------------------------------
// Postprocess + sequential time scan (one thread per batch column)
// ---------------------------------------------------------------------------
__device__ __forceinline__ void rot6d_mat(const float d[6], float R[3][3]) {
  float n1 = fmaxf(sqrtf(d[0]*d[0] + d[1]*d[1] + d[2]*d[2]), 1e-12f);
  float b1x = d[0]/n1, b1y = d[1]/n1, b1z = d[2]/n1;
  float dp = b1x*d[3] + b1y*d[4] + b1z*d[5];
  float px = d[3] - dp*b1x, py = d[4] - dp*b1y, pz = d[5] - dp*b1z;
  float n2 = fmaxf(sqrtf(px*px + py*py + pz*pz), 1e-12f);
  float b2x = px/n2, b2y = py/n2, b2z = pz/n2;
  R[0][0]=b1x; R[0][1]=b1y; R[0][2]=b1z;
  R[1][0]=b2x; R[1][1]=b2y; R[1][2]=b2z;
  R[2][0]=b1y*b2z - b1z*b2y;
  R[2][1]=b1z*b2x - b1x*b2z;
  R[2][2]=b1x*b2y - b1y*b2x;
}

__device__ __forceinline__ void mat2quat(const float R[3][3], float q[4]) {
  float m00=R[0][0], m01=R[0][1], m02=R[0][2];
  float m10=R[1][0], m11=R[1][1], m12=R[1][2];
  float m20=R[2][0], m21=R[2][1], m22=R[2][2];
  float a0 = 1.f+m00+m11+m22, a1 = 1.f+m00-m11-m22;
  float a2 = 1.f-m00+m11-m22, a3 = 1.f-m00-m11+m22;
  float s0 = sqrtf(fmaxf(a0,1e-12f)), s1 = sqrtf(fmaxf(a1,1e-12f));
  float s2 = sqrtf(fmaxf(a2,1e-12f)), s3 = sqrtf(fmaxf(a3,1e-12f));
  int idx = 0; float best = s0;
  if (s1 > best) { best = s1; idx = 1; }
  if (s2 > best) { best = s2; idx = 2; }
  if (s3 > best) { best = s3; idx = 3; }
  float c[4];
  if      (idx == 0) { c[0]=a0;      c[1]=m21-m12; c[2]=m02-m20; c[3]=m10-m01; }
  else if (idx == 1) { c[0]=m21-m12; c[1]=a1;      c[2]=m10+m01; c[3]=m02+m20; }
  else if (idx == 2) { c[0]=m02-m20; c[1]=m10+m01; c[2]=a2;      c[3]=m12+m21; }
  else               { c[0]=m10-m01; c[1]=m02+m20; c[2]=m12+m21; c[3]=a3;      }
  float inv = 1.0f / (2.0f * best);
  q[0]=c[0]*inv; q[1]=c[1]*inv; q[2]=c[2]*inv; q[3]=c[3]*inv;
}

__global__ void postproc_kernel(const float* __restrict__ traj,
                                const float* __restrict__ dec,
                                float* __restrict__ out) {
  int b = blockIdx.x * blockDim.x + threadIdx.x;
  if (b >= B_DIM) return;
  float heading = 0.0f, xy0 = 0.0f, xy1 = 0.0f;
  for (int t = 0; t < T_DIM - 1; ++t) {
    long long s = (long long)t * B_DIM + b;
    const float* d  = dec  + s * 16;
    const float* lt = traj + s * 11;
    float dlo[6] = { d[3], d[4], d[5], d[6], d[7], d[8] };
    float lo0[6] = { lt[3], lt[4], lt[5], lt[6], lt[7], lt[8] };
    float R1[3][3], R0[3][3], R[3][3];
    rot6d_mat(dlo, R1);
    rot6d_mat(lo0, R0);
    for (int i = 0; i < 3; ++i)
      for (int j = 0; j < 3; ++j)
        R[i][j] = R1[i][0]*R0[0][j] + R1[i][1]*R0[1][j] + R1[i][2]*R0[2][j];
    float ol[11];
    ol[0] = d[0]; ol[1] = d[1]; ol[2] = lt[2] + d[2];
    ol[3] = R[0][0]; ol[4] = R[0][1]; ol[5] = R[0][2];
    ol[6] = R[1][0]; ol[7] = R[1][1]; ol[8] = R[1][2];
    ol[9] = d[9]; ol[10] = d[10];
    if (t == 0) {
      const float* l0 = traj + (long long)b * 11;
      ol[0] = l0[0]; ol[1] = l0[1]; ol[9] = l0[9]; ol[10] = l0[10];
    }
    heading += atan2f(ol[10], ol[9]);
    float ch = cosf(heading), sh = sinf(heading);
    xy0 += ch * ol[0] - sh * ol[1];
    xy1 += sh * ol[0] + ch * ol[1];
    float lo2[6] = { ol[3], ol[4], ol[5], ol[6], ol[7], ol[8] };
    float Rq[3][3]; rot6d_mat(lo2, Rq);
    float ql[4]; mat2quat(Rq, ql);
    float half = 0.5f * heading;
    float hw = cosf(half), hs = sinf(half);
    float oq0 = hw*ql[0] - hs*ql[3];
    float oq1 = hw*ql[1] - hs*ql[2];
    float oq2 = hw*ql[2] + hs*ql[1];
    float oq3 = hw*ql[3] + hs*ql[0];
    float* o = out + s * 20;
    for (int i = 0; i < 11; ++i) o[i] = ol[i];
    o[11] = d[11]; o[12] = d[12];
    o[13] = xy0; o[14] = xy1; o[15] = ol[2];
    o[16] = oq0; o[17] = oq1; o[18] = oq2; o[19] = oq3;
  }
}

// ---------------------------------------------------------------------------
// Host launcher
// ---------------------------------------------------------------------------
extern "C" void kernel_launch(void* const* d_in, const int* in_sizes, int n_in,
                              void* d_out, int out_size, void* d_ws, size_t ws_size,
                              hipStream_t stream) {
  (void)in_sizes; (void)n_in; (void)out_size; (void)ws_size;
  const float* traj     = (const float*)d_in[0];
  const float* motion   = (const float*)d_in[1];
  const float* orient_q = (const float*)d_in[2];
  const float* trans    = (const float*)d_in[3];
  const float* eps      = (const float*)d_in[4];
  // post params d_in[5..14]; prior d_in[15..24] is dead code; dec d_in[25..34]
  auto F = [&](int i) { return (const float*)d_in[i]; };

  char* p = (char*)d_ws;
  auto take = [&](size_t bytes) -> char* {
    char* r = p;
    p += (bytes + 255) & ~(size_t)255;
    return r;
  };
  __bf16* xin    = (__bf16*)take((size_t)T_DIM * B_DIM * XLD * 2);  // 65536 x 544 bf16
  __bf16* zbuf   = (__bf16*)take((size_t)NROWS * 64 * 2);
  float*  qout   = (float*) take((size_t)NROWS * 96 * 4);
  float*  decout = (float*) take((size_t)NROWS * 16 * 4);
  __bf16* wp0 = (__bf16*)take((size_t)512 * 1088 * 2);
  __bf16* wp1 = (__bf16*)take((size_t)512 * 512 * 2);
  __bf16* wp2 = (__bf16*)take((size_t)512 * 512 * 2);
  __bf16* wp3 = (__bf16*)take((size_t)512 * 512 * 2);
  __bf16* wp4 = (__bf16*)take((size_t)96  * 512 * 2);
  __bf16* wd0 = (__bf16*)take((size_t)512 * 608 * 2);
  __bf16* wd1 = (__bf16*)take((size_t)512 * 512 * 2);
  __bf16* wd2 = (__bf16*)take((size_t)512 * 512 * 2);
  __bf16* wd3 = (__bf16*)take((size_t)512 * 512 * 2);
  __bf16* wd4 = (__bf16*)take((size_t)16  * 512 * 2);

  auto prep = [&](const float* src, __bf16* dst, int validK, int srcCols, int dstRows,
                  int ldDst, int kseg, int koffDst, int srcRowOff) {
    long long total = (long long)dstRows * kseg;
    int blocks = (int)((total + 255) / 256);
    prep_weight_kernel<<<blocks, 256, 0, stream>>>(src, dst, validK, srcCols, dstRows,
                                                   ldDst, kseg, koffDst, srcRowOff);
  };
  // post: W0 [1036,512] split into x_past / x_curr K-segments
  prep(F(5),  wp0, 518, 512, 512, 1088, 544, 0,   0);
  prep(F(5),  wp0, 518, 512, 512, 1088, 544, 544, 518);
  prep(F(7),  wp1, 512, 512, 512, 512, 512, 0, 0);
  prep(F(9),  wp2, 512, 512, 512, 512, 512, 0, 0);
  prep(F(11), wp3, 512, 512, 512, 512, 512, 0, 0);
  prep(F(13), wp4, 512, 96,  96,  512, 512, 0, 0);
  // dec: W0 [566,512] = x_past(518->544) + z(48->64)
  prep(F(25), wd0, 518, 512, 512, 608, 544, 0,   0);
  prep(F(25), wd0, 48,  512, 512, 608, 64,  544, 518);
  prep(F(27), wd1, 512, 512, 512, 512, 512, 0, 0);
  prep(F(29), wd2, 512, 512, 512, 512, 512, 0, 0);
  prep(F(31), wd3, 512, 512, 512, 512, 512, 0, 0);
  prep(F(33), wd4, 512, 13,  16,  512, 512, 0, 0);

  // x_in: cols 0..5 heading-frame features, 6..517 motion cast, 518..543 zero
  preprocess_kernel<<<(T_DIM * B_DIM) / 256, 256, 0, stream>>>(orient_q, trans, xin,
                                                               T_DIM * B_DIM);
  long long mtotal = (long long)T_DIM * B_DIM * M_DIM;
  cast_motion_kernel<<<(int)(mtotal / 256), 256, 0, stream>>>(motion, xin, mtotal);

  // posterior MLP (prior MLP is dead code -> skipped)
  fused_mlp_kernel<XLD, 17, 1088><<<NROWS / 64, 512, LDS_BYTES, stream>>>(
      xin, 0,                     // x_past
      xin, B_DIM,                 // x_curr = rows shifted by B
      wp0, wp1, wp2, wp3, wp4,
      F(6), F(8), F(10), F(12), F(14),
      qout, 96, 6, 96);

  long long ztotal = (long long)NROWS * 64;
  zsamp_kernel<<<(int)(ztotal / 256), 256, 0, stream>>>(qout, eps, zbuf, ztotal);

  // decoder MLP
  fused_mlp_kernel<64, 2, 608><<<NROWS / 64, 512, LDS_BYTES, stream>>>(
      xin, 0,                     // x_past
      zbuf, 0,                    // z sample (K padded 48->64)
      wd0, wd1, wd2, wd3, wd4,
      F(26), F(28), F(30), F(32), F(34),
      decout, 16, 1, 13);

  // sequential cumsum / rotation assembly
  postproc_kernel<<<4, 64, 0, stream>>>(traj, decout, (float*)d_out);
}
